// MessagePassing_50955492000253
// MI455X (gfx1250) — matrix operand order repaired and checked by reading, not compiled, not run on hardware
//
#include <hip/hip_runtime.h>

// ---------------------------------------------------------------------------
// Edge-type-gated MLP: [E*2,256] x W1[256,512] -> relu -> x W2[:, type*128:+128]
// bf16 WMMA on gfx1250 (wave32). Gather-before-GEMM2 avoids 4x wasted FLOPs.
// Round 3: W1 panels staged with GLOBAL_LOAD_ASYNC_TO_LDS_B128 (ASYNCcnt),
// double-buffered so the async DMA of panel g+1 overlaps WMMA on panel g.
// ---------------------------------------------------------------------------

typedef __attribute__((ext_vector_type(16))) __bf16 v16bf;
typedef __attribute__((ext_vector_type(8)))  float  v8f;

union BF16x16 { v16bf v; unsigned u[8]; uint4 q[2]; };

#define E_TOTAL 131072
#define WIDTH   128
#define IN_CH   256
#define MID     512
#define OUTC    128
#define NT      8
#define OUTW    (OUTC * NT)   // 1024

#define EB      64            // edges per block
#define ROWS    (EB * 2)      // 128 rows (edge x dim)
#define THREADS 256
#define AS      (IN_CH + 8)   // A tile LDS stride (bf16 elems), 264 (8-elem aligned)
#define PK      (IN_CH + 8)   // W1 panel K-stride (col-major), 264
#define HS      (MID + 8)     // H tile LDS stride, 520

#define A_BYTES   (ROWS * AS * 2)       // 67584
#define P1_BYTES  (64 * PK * 2)         // 33792 (one panel buffer)
#define P_BYTES   (2 * P1_BYTES)        // double-buffered
#define H_BYTES   (ROWS * HS * 2)       // 133120
#define CNT_BYTES (NT * 4)              // 32
#define LIST_BYTES (NT * ROWS * 4)      // 4096
#define SMEM_BYTES (A_BYTES + P_BYTES + H_BYTES + CNT_BYTES + LIST_BYTES) // 272416

// workspace layout (bf16 col-major weights)
#define WS_W1_ELEMS (MID * IN_CH)       // [512 cols][256 k]
#define WS_W2_ELEMS (OUTW * MID)        // [1024 cols][512 k]
#define WS_NEEDED   ((size_t)(WS_W1_ELEMS + WS_W2_ELEMS) * 2)   // 1,310,720 B

__device__ __forceinline__ unsigned short f2bf(float f) {
  unsigned u = __float_as_uint(f);
  u += 0x7FFFu + ((u >> 16) & 1u);     // round-to-nearest-even
  return (unsigned short)(u >> 16);
}
__device__ __forceinline__ unsigned pack_bf2(float a, float b) {
  return (unsigned)f2bf(a) | ((unsigned)f2bf(b) << 16);
}

// gfx1250 async global->LDS 16B copy (per-lane addresses), ASYNCcnt-tracked.
__device__ __forceinline__ void async_b128_g2l(unsigned lds_byte_addr, const void* gptr) {
  asm volatile("global_load_async_to_lds_b128 %0, %1, off"
               :
               : "v"(lds_byte_addr), "v"((unsigned long long)(uintptr_t)gptr)
               : "memory");
}
__device__ __forceinline__ void wait_async0() {
  asm volatile("s_wait_asynccnt 0x0" ::: "memory");
}

// ---- one-shot weight transform: f32 row-major -> bf16 col-major ------------
__global__ __launch_bounds__(THREADS)
void preconv_kernel(const float* __restrict__ W1, const float* __restrict__ W2,
                    unsigned short* __restrict__ w1b, unsigned short* __restrict__ w2b) {
  int id = blockIdx.x * THREADS + threadIdx.x;
  if (id < WS_W1_ELEMS) {                       // W1[k][n] -> w1b[n][k]
    int k = id / MID, n = id % MID;
    w1b[(size_t)n * IN_CH + k] = f2bf(W1[id]);
  } else {
    int id2 = id - WS_W1_ELEMS;
    if (id2 < WS_W2_ELEMS) {                    // W2[k][n] -> w2b[n][k]
      int k = id2 / OUTW, n = id2 % OUTW;
      w2b[(size_t)n * MID + k] = f2bf(W2[id2]);
    }
  }
}

template <bool PRE>
__global__ __launch_bounds__(THREADS)
void edge_mlp_kernel(const float* __restrict__ fromE,
                     const float* __restrict__ toE,
                     const int*   __restrict__ etype,
                     const float* __restrict__ W1,
                     const float* __restrict__ W2,
                     const unsigned short* __restrict__ w1b,
                     const unsigned short* __restrict__ w2b,
                     float*       __restrict__ out) {
  extern __shared__ char smem[];
  unsigned short* aT = (unsigned short*)smem;                               // [ROWS][AS]
  unsigned short* pT = (unsigned short*)(smem + A_BYTES);                   // 2x [64][PK]
  unsigned short* hT = (unsigned short*)(smem + A_BYTES + P_BYTES);         // [ROWS][HS]
  int* counts  = (int*)(smem + A_BYTES + P_BYTES + H_BYTES);                // [NT]
  int* rowlist = counts + NT;                                               // [NT][ROWS]

  const int tid    = threadIdx.x;
  const int wave   = tid >> 5;
  const int lane   = tid & 31;
  const int laneLo = lane & 15;
  const int laneHi = (lane >> 4) & 1;
  const int e0     = blockIdx.x * EB;
  // low 32 bits of a flat LDS address == LDS byte offset (ISA 10.2 aperture rule)
  const unsigned ldsPanelBase = (unsigned)(uintptr_t)(smem + A_BYTES);

  // issue one panel's async DMA: 64 cols x 256 k bf16 = 2048 x 16B chunks
  auto issuePanel = [&](int g, int buf) {
    const unsigned base = ldsPanelBase + (unsigned)buf * P1_BYTES;
    #pragma unroll
    for (int it = 0; it < 8; ++it) {
      int id = it * THREADS + tid;
      int c  = id >> 5;                           // 0..63
      int k  = (id & 31) << 3;                    // 0..255 step 8
      async_b128_g2l(base + (unsigned)(c * PK + k) * 2,
                     &w1b[(size_t)(g * 64 + c) * IN_CH + k]);
    }
  };

  if constexpr (PRE) issuePanel(0, 0);            // overlap with phase 0

  // ---------------- Phase 0: stage A tile (f32 -> bf16) into LDS ----------
  #pragma unroll
  for (int it = 0; it < (ROWS * IN_CH / 4) / THREADS; ++it) {   // 32 iters
    int q  = it * THREADS + tid;
    int r  = q >> 6;                   // 64 float4 quads per row
    int k  = (q & 63) << 2;
    int eb = r >> 1, d = r & 1;
    const float* src = (k < WIDTH)
        ? (fromE + (size_t)(e0 + eb) * IN_CH + d * WIDTH + k)
        : (toE   + (size_t)(e0 + eb) * IN_CH + d * WIDTH + (k - WIDTH));
    float4 f = *(const float4*)src;
    unsigned* dst = (unsigned*)&aT[r * AS + k];
    dst[0] = pack_bf2(f.x, f.y);
    dst[1] = pack_bf2(f.z, f.w);
  }
  __syncthreads();

  // ---------------- Phase 1: GEMM1 (A[128x256] x W1[256x512]) + ReLU ------
  for (int g = 0; g < MID / 64; ++g) {            // 8 x 64-col W1 panels
    unsigned short* pb = pT + (size_t)(g & 1) * (64 * PK);
    if constexpr (PRE) {
      wait_async0();                              // panel g resident in LDS
      __syncthreads();
      if (g + 1 < MID / 64) issuePanel(g + 1, (g + 1) & 1);  // overlap w/ WMMA
    } else {
      int c  = tid & 63;
      int k0 = tid >> 6;                          // 0..3
      for (int k = k0; k < IN_CH; k += 4)
        pb[c * PK + k] = f2bf(W1[(size_t)k * MID + g * 64 + c]);
      __syncthreads();
    }

    const int mrow = wave * 16;
    v8f acc[4];
    #pragma unroll
    for (int n = 0; n < 4; ++n)
      #pragma unroll
      for (int i = 0; i < 8; ++i) acc[n][i] = 0.f;

    #pragma unroll
    for (int kt = 0; kt < IN_CH / 32; ++kt) {     // 8 K-steps
      BF16x16 a;
      const int rowA = mrow + laneLo;
      const int kbA  = kt * 32 + laneHi * 16;     // A: one 32B K-chunk
      a.q[0] = *(const uint4*)&aT[rowA * AS + kbA];
      a.q[1] = *(const uint4*)&aT[rowA * AS + kbA + 8];
      #pragma unroll
      for (int n = 0; n < 4; ++n) {
        BF16x16 b;
        const int c  = n * 16 + laneLo;
        const int kbB = kt * 32 + laneHi * 16;    // B: one 32B K-chunk
        b.q[0] = *(const uint4*)&pb[c * PK + kbB];
        b.q[1] = *(const uint4*)&pb[c * PK + kbB + 8];
        acc[n] = __builtin_amdgcn_wmma_f32_16x16x32_bf16(
            false, a.v, false, b.v, (short)0, acc[n], false, false);
      }
    }
    // ReLU + bf16 pack into H tile (C layout: VGPR i -> M = i + 8*laneHi)
    #pragma unroll
    for (int n = 0; n < 4; ++n)
      #pragma unroll
      for (int i = 0; i < 8; ++i) {
        int r   = mrow + i + laneHi * 8;
        int col = g * 64 + n * 16 + laneLo;
        hT[r * HS + col] = f2bf(fmaxf(acc[n][i], 0.f));
      }
    if constexpr (!PRE) __syncthreads();
  }

  // ---------------- Phase 2: bucket rows by edge type ---------------------
  __syncthreads();
  if (tid < NT) counts[tid] = 0;
  __syncthreads();
  if (tid < EB) {
    int t   = etype[e0 + tid];
    int pos = atomicAdd(&counts[t], 2);           // both dims share the type
    rowlist[t * ROWS + pos]     = tid * 2;
    rowlist[t * ROWS + pos + 1] = tid * 2 + 1;
  }
  __syncthreads();

  // ---------------- Phase 3: GEMM2 on compacted same-type chunks ----------
  int task = 0;
  for (int t = 0; t < NT; ++t) {
    const int cnt = counts[t];
    const int nch = (cnt + 15) >> 4;
    for (int c = 0; c < nch; ++c, ++task) {
      if ((task & 7) != wave) continue;           // round-robin chunks
      const int base = c * 16;
      const int ja   = base + laneLo;
      const int rowA = rowlist[t * ROWS + (ja < cnt ? ja : base)];

      for (int half = 0; half < 2; ++half) {      // 2 x 64 output cols
        v8f acc[4];
        #pragma unroll
        for (int n = 0; n < 4; ++n)
          #pragma unroll
          for (int i = 0; i < 8; ++i) acc[n][i] = 0.f;

        for (int kt = 0; kt < MID / 32; ++kt) {   // 16 K-steps
          BF16x16 a;
          const int kbA = kt * 32 + laneHi * 16;
          a.q[0] = *(const uint4*)&hT[rowA * HS + kbA];
          a.q[1] = *(const uint4*)&hT[rowA * HS + kbA + 8];
          #pragma unroll
          for (int n = 0; n < 4; ++n) {
            BF16x16 b;
            const int colW2 = t * OUTC + half * 64 + n * 16 + laneLo;
            if constexpr (PRE) {
              const int kbB = kt * 32 + laneHi * 16;
              b.q[0] = *(const uint4*)&w2b[(size_t)colW2 * MID + kbB];
              b.q[1] = *(const uint4*)&w2b[(size_t)colW2 * MID + kbB + 8];
            } else {
              #pragma unroll
              for (int v = 0; v < 8; ++v) {
                int kk = kt * 32 + laneHi * 16 + v * 2;
                b.u[v] = pack_bf2(W2[(size_t)kk * OUTW + colW2],
                                  W2[(size_t)(kk + 1) * OUTW + colW2]);
              }
            }
            acc[n] = __builtin_amdgcn_wmma_f32_16x16x32_bf16(
                false, a.v, false, b.v, (short)0, acc[n], false, false);
          }
        }
        // scatter to output rows (tail-masked)
        #pragma unroll
        for (int i = 0; i < 8; ++i) {
          int M = i + laneHi * 8;
          if (base + M < cnt) {
            int row  = rowlist[t * ROWS + base + M];
            size_t o = (size_t)(blockIdx.x * ROWS + row) * OUTC
                     + half * 64 + laneLo;
            #pragma unroll
            for (int n = 0; n < 4; ++n)
              out[o + (size_t)n * 16] = acc[n][i];
          }
        }
      }
    }
  }
}

extern "C" void kernel_launch(void* const* d_in, const int* in_sizes, int n_in,
                              void* d_out, int out_size, void* d_ws, size_t ws_size,
                              hipStream_t stream) {
  (void)in_sizes; (void)n_in; (void)out_size;
  const float* fromE = (const float*)d_in[0];
  const float* toE   = (const float*)d_in[1];
  const int*   etype = (const int*)d_in[2];
  const float* W1    = (const float*)d_in[3];
  const float* W2    = (const float*)d_in[4];
  float*       out   = (float*)d_out;

  dim3 grid(E_TOTAL / EB);   // 2048 blocks
  dim3 block(THREADS);

  const bool pre = (d_ws != nullptr) && (ws_size >= WS_NEEDED);
  if (pre) {
    unsigned short* w1b = (unsigned short*)d_ws;
    unsigned short* w2b = w1b + WS_W1_ELEMS;
    int total = WS_W1_ELEMS + WS_W2_ELEMS;
    preconv_kernel<<<(total + THREADS - 1) / THREADS, THREADS, 0, stream>>>(
        W1, W2, w1b, w2b);
    (void)hipFuncSetAttribute(
        reinterpret_cast<const void*>(&edge_mlp_kernel<true>),
        hipFuncAttributeMaxDynamicSharedMemorySize, SMEM_BYTES);
    edge_mlp_kernel<true><<<grid, block, SMEM_BYTES, stream>>>(
        fromE, toE, etype, W1, W2, w1b, w2b, out);
  } else {
    (void)hipFuncSetAttribute(
        reinterpret_cast<const void*>(&edge_mlp_kernel<false>),
        hipFuncAttributeMaxDynamicSharedMemorySize, SMEM_BYTES);
    edge_mlp_kernel<false><<<grid, block, SMEM_BYTES, stream>>>(
        fromE, toE, etype, W1, W2, nullptr, nullptr, out);
  }
}
